// ScaleDotProductAttention_47158740910475
// MI455X (gfx1250) — matrix operand order repaired and checked by reading, not compiled, not run on hardware
//
#include <hip/hip_runtime.h>

// Problem constants (B, H, S, D) from the reference
#define B_ 2
#define H_ 16
#define S_ 2048
#define D_ 128
#define SCALE 0.08838834764831845f   // 1/sqrt(128)
#define ROWS_PER_WAVE 32             // two 16-row WMMA tiles per wave: K/V fragments reused 2x
#define WAVES_PER_BLOCK 4            // 128-thread blocks, no LDS -> occupancy is VGPR-limited

typedef __attribute__((ext_vector_type(16))) _Float16 v16h;
typedef __attribute__((ext_vector_type(8)))  float    v8f;
typedef __attribute__((ext_vector_type(4)))  float    v4f;

// 16-bit A-matrix 16x32 (MxK) layout, ISA 7.12.2:
// lanes 0-15 hold K = {0..7, 16..23}; lanes 16-31 hold K = {8..15, 24..31}
__device__ __forceinline__ int a_frag_k(int lane, int e) {
    int kbase = (lane < 16) ? 0 : 8;
    int vg = e >> 1, w = e & 1;
    return (vg < 4) ? (kbase + 2 * vg + w) : (kbase + 16 + 2 * (vg - 4) + w);
}

// 16-bit B-matrix 32x16 (KxN) layout: lanes 0-15 hold K = {0..7, 16..23}; lanes 16-31 hold K = {8..15, 24..31}
__device__ __forceinline__ int b_frag_k(int lane, int e) {
    return ((lane < 16) ? 0 : 8) + ((e < 8) ? e : (8 + e));
}

#define WMMA_F16(A, B, C) \
    __builtin_amdgcn_wmma_f32_16x16x32_f16(false, (A), false, (B), (short)0, (C), false, false)

extern "C" __global__ __launch_bounds__(32 * WAVES_PER_BLOCK, 1)
void attn_fused_wmma_kernel(const float* __restrict__ Q, const float* __restrict__ Km,
                            const float* __restrict__ V, const int* __restrict__ Mask,
                            float* __restrict__ Out, float* Score) {
    const int lane = threadIdx.x & 31;
    const int wave = threadIdx.x >> 5;

    const int mblocks  = S_ / ROWS_PER_WAVE;                        // 64 row-blocks per (b,h)
    const int rowblock = blockIdx.x * WAVES_PER_BLOCK + wave;       // global 32-row block
    const int bh = rowblock / mblocks;
    const int m0 = (rowblock % mblocks) * ROWS_PER_WAVE;
    const int b  = bh / H_;

    const float* Qbh = Q  + (size_t)bh * S_ * D_;
    const float* Kbh = Km + (size_t)bh * S_ * D_;
    const float* Vbh = V  + (size_t)bh * S_ * D_;
    const int*   Mb  = Mask + (size_t)b * S_ * S_;                  // mask shared across H: L2-reused
    float* Obh = Out   + (size_t)bh * S_ * D_;
    float* Sbh = Score + (size_t)bh * S_ * S_;                      // also our L2-resident scratch

    const int l16 = lane & 15;

    // ---------------- load Q: two 16-row tiles, 4 K32 chunks each, fold 1/sqrt(D)
    v16h qa[2][4];
    #pragma unroll
    for (int t = 0; t < 2; ++t)
        #pragma unroll
        for (int c = 0; c < 4; ++c)
            #pragma unroll
            for (int e = 0; e < 16; ++e) {
                int d = c * 32 + a_frag_k(lane, e);
                qa[t][c][e] = (_Float16)(Qbh[(size_t)(m0 + t * 16 + l16) * D_ + d] * SCALE);
            }

    // ---------------- pass 1: scores = (Q*scale) @ K^T, masked, -> Score (stays hot in 192MB L2)
    for (int nt = 0; nt < S_ / 16; ++nt) {
        v8f acc0 = {}, acc1 = {};
        #pragma unroll
        for (int c = 0; c < 4; ++c) {
            v16h kb;   // B fragment: B[kd][n] = K[nt*16 + n][c*32 + kd]; reused by both row tiles
            #pragma unroll
            for (int e = 0; e < 16; ++e)
                kb[e] = (_Float16)Kbh[(size_t)(nt * 16 + l16) * D_ + c * 32 + b_frag_k(lane, e)];
            acc0 = WMMA_F16(qa[0][c], kb, acc0);
            acc1 = WMMA_F16(qa[1][c], kb, acc1);
        }
        const int col   = nt * 16 + l16;
        const int rbase = (lane < 16) ? 0 : 8;
        #pragma unroll
        for (int r = 0; r < 8; ++r) {
            int row0 = rbase + r, row1 = 16 + rbase + r;
            float s0 = acc0[r], s1 = acc1[r];
            s0 = (Mb[(size_t)(m0 + row0) * S_ + col] == 0) ? 1e-9f : s0;
            s1 = (Mb[(size_t)(m0 + row1) * S_ + col] == 0) ? 1e-9f : s1;
            Sbh[(size_t)(m0 + row0) * S_ + col] = s0;
            Sbh[(size_t)(m0 + row1) * S_ + col] = s1;
        }
    }
    asm volatile("s_wait_storecnt 0x0" ::: "memory");   // scores visible before re-read

    // ---------------- pass 2: exact softmax, one row at a time, lanes split columns (coalesced)
    for (int row = 0; row < ROWS_PER_WAVE; ++row) {
        v4f* base4 = (v4f*)(Sbh + (size_t)(m0 + row) * S_);   // 512 float4 per row
        v4f rv[16];
        #pragma unroll
        for (int j = 0; j < 16; ++j) rv[j] = base4[j * 32 + lane];
        float mx = -3.402823466e38f;
        #pragma unroll
        for (int j = 0; j < 16; ++j) {
            mx = fmaxf(mx, fmaxf(fmaxf(rv[j].x, rv[j].y), fmaxf(rv[j].z, rv[j].w)));
        }
        #pragma unroll
        for (int off = 16; off >= 1; off >>= 1) mx = fmaxf(mx, __shfl_xor(mx, off, 32));
        float sum = 0.f;
        #pragma unroll
        for (int j = 0; j < 16; ++j) {
            rv[j].x = __expf(rv[j].x - mx); rv[j].y = __expf(rv[j].y - mx);
            rv[j].z = __expf(rv[j].z - mx); rv[j].w = __expf(rv[j].w - mx);
            sum += rv[j].x + rv[j].y + rv[j].z + rv[j].w;
        }
        #pragma unroll
        for (int off = 16; off >= 1; off >>= 1) sum += __shfl_xor(sum, off, 32);
        const float inv = 1.0f / sum;
        #pragma unroll
        for (int j = 0; j < 16; ++j) base4[j * 32 + lane] = rv[j] * inv;
    }
    asm volatile("s_wait_storecnt 0x0" ::: "memory");   // probabilities visible before P@V

    // ---------------- pass 3: Out = P @ V (two 16x128 tiles; V fragments reused 2x)
    v8f oacc0[8] = {}, oacc1[8] = {};
    for (int kc = 0; kc < S_ / 32; ++kc) {
        v16h pa0, pa1;   // A fragments from L2-resident probabilities
        #pragma unroll
        for (int e = 0; e < 16; ++e) {
            int n = kc * 32 + a_frag_k(lane, e);
            pa0[e] = (_Float16)Sbh[(size_t)(m0 + l16) * S_ + n];
            pa1[e] = (_Float16)Sbh[(size_t)(m0 + 16 + l16) * S_ + n];
        }
        #pragma unroll
        for (int dt = 0; dt < 8; ++dt) {
            v16h vb;  // B fragment: B[n][d] = V[kc*32 + n][dt*16 + d]
            #pragma unroll
            for (int e = 0; e < 16; ++e)
                vb[e] = (_Float16)Vbh[(size_t)(kc * 32 + b_frag_k(lane, e)) * D_ + dt * 16 + l16];
            oacc0[dt] = WMMA_F16(pa0, vb, oacc0[dt]);
            oacc1[dt] = WMMA_F16(pa1, vb, oacc1[dt]);
        }
    }

    // ---------------- store Out (streamed, never re-read -> nontemporal)
    {
        const int rbase = (lane < 16) ? 0 : 8;
        #pragma unroll
        for (int dt = 0; dt < 8; ++dt)
            #pragma unroll
            for (int r = 0; r < 8; ++r) {
                __builtin_nontemporal_store(
                    oacc0[dt][r], &Obh[(size_t)(m0 + rbase + r) * D_ + dt * 16 + l16]);
                __builtin_nontemporal_store(
                    oacc1[dt][r], &Obh[(size_t)(m0 + 16 + rbase + r) * D_ + dt * 16 + l16]);
            }
    }
}

extern "C" void kernel_launch(void* const* d_in, const int* in_sizes, int n_in,
                              void* d_out, int out_size, void* d_ws, size_t ws_size,
                              hipStream_t stream) {
    const float* q    = (const float*)d_in[0];
    const float* k    = (const float*)d_in[1];
    const float* v    = (const float*)d_in[2];
    const int*   mask = (const int*)d_in[3];

    float* out   = (float*)d_out;                       // (B,H,S,D)
    float* score = out + (size_t)B_ * H_ * S_ * D_;     // (B,H,S,S)

    const int waves  = B_ * H_ * (S_ / ROWS_PER_WAVE);  // 2048 waves
    const int blocks = waves / WAVES_PER_BLOCK;         // 512 workgroups x 128 threads

    hipLaunchKernelGGL(attn_fused_wmma_kernel, dim3(blocks), dim3(32 * WAVES_PER_BLOCK),
                       0, stream, q, k, v, mask, out, score);
}
